// Gate_28329604284810
// MI455X (gfx1250) — matrix operand order repaired and checked by reading, not compiled, not run on hardware
//
#include <hip/hip_runtime.h>

// MoE sigmoid router: fused [16384,7168]x[7168,256]^T GEMM (bf16 WMMA, f32 acc)
// + grouped top-k routing, for MI455X (gfx1250, wave32).
//
// Roofline: 60.1 GFLOP GEMM / 478 MB HBM traffic -> 20.5us floor @ 23.3 TB/s;
// needs ~2.9 PFLOP/s effective -> only the bf16/fp8 WMMA pipes reach that.
// x streams once (NT hint + speculative prefetch), weight (7.3 MB) stays
// L2-resident (RT). Logits tile never leaves LDS (routing fused).

typedef __attribute__((ext_vector_type(16))) __bf16 v16bf;
typedef __attribute__((ext_vector_type(8)))  float  v8f;
typedef __attribute__((ext_vector_type(4)))  float  f32x4;

#define T_ROWS   16384
#define DIM      7168
#define NEXP     256
#define TILE_M   16
#define KSTEP    32
#define N_GROUPS 8
#define TOP_K    8

// sum of the two largest values across the wave (order-invariant, tie-safe)
__device__ __forceinline__ float warp_top2_sum(float v) {
    float m1 = v, m2 = -INFINITY;
#pragma unroll
    for (int off = 16; off; off >>= 1) {
        float o1 = __shfl_xor(m1, off, 32);
        float o2 = __shfl_xor(m2, off, 32);
        float hi = fmaxf(m1, o1);
        float lo = fminf(m1, o1);
        m1 = hi;
        m2 = fmaxf(lo, fmaxf(m2, o2));
    }
    return m1 + m2;
}

// max value, ties broken toward LOWEST index (matches jax.lax.top_k ordering)
__device__ __forceinline__ void warp_argmax(float& v, int& i) {
#pragma unroll
    for (int off = 16; off; off >>= 1) {
        float ov = __shfl_xor(v, off, 32);
        int   oi = __shfl_xor(i, off, 32);
        if (ov > v || (ov == v && oi < i)) { v = ov; i = oi; }
    }
}

__device__ __forceinline__ v16bf cvt16(f32x4 a, f32x4 b, f32x4 c, f32x4 d) {
    v16bf r;
    r[0]  = (__bf16)a.x; r[1]  = (__bf16)a.y; r[2]  = (__bf16)a.z; r[3]  = (__bf16)a.w;
    r[4]  = (__bf16)b.x; r[5]  = (__bf16)b.y; r[6]  = (__bf16)b.z; r[7]  = (__bf16)b.w;
    r[8]  = (__bf16)c.x; r[9]  = (__bf16)c.y; r[10] = (__bf16)c.z; r[11] = (__bf16)c.w;
    r[12] = (__bf16)d.x; r[13] = (__bf16)d.y; r[14] = (__bf16)d.z; r[15] = (__bf16)d.w;
    return r;
}

// fast sigmoid: v_exp_f32 + v_rcp_f32 (skip IEEE div fixup sequence)
__device__ __forceinline__ float fast_sigmoid(float v) {
    return __builtin_amdgcn_rcpf(1.0f + __expf(-v));
}

// A fragment: 16x32 bf16 (lane<16 -> K 0..15 of row M=lane,
// lane>=16 -> K 16..31). x is streamed once -> non-temporal.
__device__ __forceinline__ v16bf load_a(const float* ap) {
    f32x4 a0 = __builtin_nontemporal_load((const f32x4*)(ap + 0));
    f32x4 a1 = __builtin_nontemporal_load((const f32x4*)(ap + 4));
    f32x4 a2 = __builtin_nontemporal_load((const f32x4*)(ap + 8));
    f32x4 a3 = __builtin_nontemporal_load((const f32x4*)(ap + 12));
    return cvt16(a0, a1, a2, a3);
}

// B fragment: 32x16 bf16 (lane<16 -> K 0..15 of col N=lane, lane>=16 -> K 16..31).
// Column N = expert row -> 16 contiguous floats. Keep RT (L2-resident weight).
__device__ __forceinline__ v16bf load_b(const float* bp) {
    f32x4 b0 = *(const f32x4*)(bp + 0);
    f32x4 b1 = *(const f32x4*)(bp + 4);
    f32x4 b2 = *(const f32x4*)(bp + 8);
    f32x4 b3 = *(const f32x4*)(bp + 12);
    return cvt16(b0, b1, b2, b3);
}

__global__ __launch_bounds__(256)
void gate_fused_kernel(const float* __restrict__ x,
                       const float* __restrict__ wgt,
                       const float* __restrict__ bias,
                       float* __restrict__ out_w,
                       int*   __restrict__ out_i) {
    __shared__ float s_orig[TILE_M][NEXP];   // sigmoid scores
    __shared__ float s_bias[TILE_M][NEXP];   // sigmoid + bias

    const int wave  = threadIdx.x >> 5;      // 0..7
    const int lane  = threadIdx.x & 31;
    const int l16   = lane & 15;
    const int lhalf = lane >> 4;             // 0 or 1
    const int row0  = blockIdx.x * TILE_M;

    // ---------------- Phase 1: GEMM tile (16 rows x 32 experts per wave) ----
    v8f acc0 = {};
    v8f acc1 = {};
    const int    e0    = wave * 32;
    const size_t xrow  = (size_t)(row0 + l16) * DIM;
    const size_t wrow0 = (size_t)(e0 + l16) * DIM;
    const size_t wrow1 = (size_t)(e0 + 16 + l16) * DIM;

    const float* apb  = x   + xrow  + lhalf * 16;
    const float* bpb0 = wgt + wrow0 + lhalf * 16;
    const float* bpb1 = wgt + wrow1 + lhalf * 16;

#pragma unroll 4
    for (int kb = 0; kb < DIM; kb += KSTEP) {
        // pull the x stream ahead (global_prefetch_b8, speculative -> safe OOB)
        __builtin_prefetch((const void*)(apb + kb + 8 * KSTEP), 0, 0);

        v16bf A  = load_a(apb + kb);
        v16bf B0 = load_b(bpb0 + kb);
        v16bf B1 = load_b(bpb1 + kb);

        acc0 = __builtin_amdgcn_wmma_f32_16x16x32_bf16(
            false, A, false, B0, (short)0, acc0, false, false);
        acc1 = __builtin_amdgcn_wmma_f32_16x16x32_bf16(
            false, A, false, B1, (short)0, acc1, false, false);
    }

    // sigmoid + bias, spill logits tile into LDS
    const float bval0 = bias[e0 + l16];
    const float bval1 = bias[e0 + 16 + l16];
#pragma unroll
    for (int r = 0; r < 8; ++r) {
        const int m = r + lhalf * 8;   // C layout: VGPR r = row r (lanes 0-15) / r+8
        float sig0 = fast_sigmoid(acc0[r]);
        float sig1 = fast_sigmoid(acc1[r]);
        s_orig[m][e0 + l16]      = sig0;
        s_bias[m][e0 + l16]      = sig0 + bval0;
        s_orig[m][e0 + 16 + l16] = sig1;
        s_bias[m][e0 + 16 + l16] = sig1 + bval1;
    }
    __syncthreads();

    // ---------------- Phase 2: routing, one wave per row --------------------
#pragma unroll
    for (int rr = 0; rr < 2; ++rr) {
        const int m    = wave + rr * 8;
        const int grow = row0 + m;

        // group g == experts [g*32, g*32+32) == exactly the 32 lanes
        float sb[N_GROUPS];
#pragma unroll
        for (int g = 0; g < N_GROUPS; ++g) sb[g] = s_bias[m][g * 32 + lane];

        // group score = sum of top-2 biased scores in group (merge butterfly)
        float gs[N_GROUPS];
#pragma unroll
        for (int g = 0; g < N_GROUPS; ++g) gs[g] = warp_top2_sum(sb[g]);

        // top-4 groups by rank (ties -> lower group index), mask others to 0.0
        float sm[N_GROUPS];
#pragma unroll
        for (int g = 0; g < N_GROUPS; ++g) {
            int rank = 0;
#pragma unroll
            for (int g2 = 0; g2 < N_GROUPS; ++g2)
                rank += (gs[g2] > gs[g]) || (gs[g2] == gs[g] && g2 < g);
            sm[g] = (rank < 4) ? sb[g] : 0.0f;         // multiplicative mask -> 0
        }

        // top-8 experts over the 256 masked scores
        int   myidx = 0;
        float wsum  = 0.0f;
#pragma unroll
        for (int k = 0; k < TOP_K; ++k) {
            float bv = -INFINITY; int bi = 0x7FFFFFFF;
#pragma unroll
            for (int g = 0; g < N_GROUPS; ++g) {
                int ei = g * 32 + lane;
                if (sm[g] > bv || (sm[g] == bv && ei < bi)) { bv = sm[g]; bi = ei; }
            }
            warp_argmax(bv, bi);                       // all lanes agree on bi
            if (lane == k) myidx = bi;
            wsum += s_orig[m][bi];                     // broadcast LDS read
            const int bg = bi >> 5, bl = bi & 31;
#pragma unroll
            for (int g = 0; g < N_GROUPS; ++g)
                if (g == bg && lane == bl) sm[g] = -INFINITY;  // remove winner
        }

        if (lane < TOP_K) {
            float wt = s_orig[m][myidx] * __builtin_amdgcn_rcpf(wsum) * 2.5f;
            out_w[(size_t)grow * TOP_K + lane] = wt;
            out_i[(size_t)grow * TOP_K + lane] = myidx;
        }
    }
}

extern "C" void kernel_launch(void* const* d_in, const int* in_sizes, int n_in,
                              void* d_out, int out_size, void* d_ws, size_t ws_size,
                              hipStream_t stream) {
    const float* x    = (const float*)d_in[0];   // [16384, 7168]
    const float* wgt  = (const float*)d_in[1];   // [256, 7168]
    const float* bias = (const float*)d_in[2];   // [256]

    float* out_w = (float*)d_out;                                  // [16384, 8] f32
    int*   out_i = (int*)((float*)d_out + (size_t)T_ROWS * TOP_K); // [16384, 8] i32

    dim3 grid(T_ROWS / TILE_M);   // 1024 blocks, 16 rows each
    dim3 block(256);              // 8 waves (wave32)
    gate_fused_kernel<<<grid, block, 0, stream>>>(x, wgt, bias, out_w, out_i);
}
